// ContrastiveLoss_15522011808350
// MI455X (gfx1250) — matrix operand order repaired and checked by reading, not compiled, not run on hardware
//
#include <hip/hip_runtime.h>

// Supervised contrastive loss, fused GEMM + masked softmax-sums. N=16384, D=128.
// MI455X plan:
//   - Pre-split fp32 -> f16 hi + f16 lo planes in workspace (8 MB, L2-resident).
//   - dot = hi*hi + hi*lo + lo*hi via 3x v_wmma_f32_16x16x32_f16 per K=32 chunk
//     (fp32-class accuracy on the fast XDL f16 pipe).
//   - B tiles staged by the Tensor Data Mover (tensor_load_to_lds) with LDS row
//     padding (256B row + 32B pad = 288B pitch, 32B aligned, bank-staggered),
//     double-buffered, wave0 issues + s_wait_tensorcnt gates a block barrier.
//   - Branchless epilogue: v_exp_f32 with 1/T*log2e folded in, v_cndmask masks.
//   - Per-row sums reduced with 16-lane shfl_xor (wave32); partials written to
//     unique workspace slots (no atomics -> bit-deterministic across replays).

typedef __attribute__((ext_vector_type(16))) _Float16 v16h;
typedef __attribute__((ext_vector_type(8)))  _Float16 v8h;
typedef __attribute__((ext_vector_type(8)))  float    v8f;
typedef __attribute__((ext_vector_type(4)))  unsigned int u32x4;
typedef __attribute__((ext_vector_type(8)))  int      i32x8;
typedef __attribute__((ext_vector_type(4)))  int      i32x4;

#define DIM 128
#define TW 16
#define WAVES 8
#define ROWS_PER_BLOCK (TW * WAVES)       // 128
#define LDS_STRIDE 144                    // halves: 288B pitch (256B row + 32B TDM pad)
#define PLANE (TW * LDS_STRIDE)           // halves per tile plane (4608 B)
#define SCALE_EXP2 2.8853900817779268f    // (1/0.5) * log2(e):  exp(sim/T) = exp2(SCALE*sim)

// ---------------------------------------------------------------- prep: split
__global__ __launch_bounds__(256) void prep_split(
    const float* __restrict__ src, _Float16* __restrict__ hi,
    _Float16* __restrict__ lo, int nVec8)
{
    const int idx = blockIdx.x * 256 + threadIdx.x;
    if (idx >= nVec8) return;
    const float4* s = ((const float4*)src) + (size_t)idx * 2;
    float4 f0 = s[0], f1 = s[1];
    float xs[8] = {f0.x, f0.y, f0.z, f0.w, f1.x, f1.y, f1.z, f1.w};
    v8h hv, lv;
#pragma unroll
    for (int q = 0; q < 8; ++q) {
        _Float16 h = (_Float16)xs[q];
        hv[q] = h;
        lv[q] = (_Float16)(xs[q] - (float)h);
    }
    ((v8h*)hi)[idx] = hv;
    ((v8h*)lo)[idx] = lv;
}

// ------------------------------------------------- TDM: stage one 16x128 tile
__device__ __forceinline__ void tdm_load_tile(const _Float16* gsrc, unsigned int ldsAddr) {
    const unsigned long long ga = (unsigned long long)(uintptr_t)gsrc;
    u32x4 g0;
    g0[0] = 1u;                                   // count=1, user descriptor
    g0[1] = ldsAddr;                              // LDS byte address
    g0[2] = (unsigned int)(ga & 0xFFFFFFFFu);
    g0[3] = (unsigned int)((ga >> 32) & 0x1FFFFFFu) | (2u << 30);  // type=2 (image)
    i32x8 g1;
    g1[0] = (1 << 16)        // data_size = 2 bytes
          | (1 << 20)        // pad_enable
          | (5 << 22)        // pad_interval: 64 DWORDs (=256B row)
          | (7 << 25);       // pad_amount:   8 DWORDs (=32B)  -> 288B LDS pitch
    g1[1] = (DIM << 16);                          // tensor_dim0 = 128
    g1[2] = (TW << 16);                           // tensor_dim1 = 16 (tile-local)
    g1[3] = (DIM << 16);                          // tile_dim0 = 128
    g1[4] = TW;                                   // tile_dim1 = 16, tile_dim2 = 0
    g1[5] = DIM;                                  // tensor_dim0_stride = 128
    g1[6] = 0;
    g1[7] = 0;
    i32x4 z4 = {};
    i32x8 z8 = {};
    __builtin_amdgcn_tensor_load_to_lds(g0, g1, z4, z4, z8, 0);
}

// --------------------------------------------------------- main (TDM staging)
__global__ __launch_bounds__(256) void contrastive_tdm(
    const _Float16* __restrict__ Ehi, const _Float16* __restrict__ Elo,
    const int* __restrict__ labels,
    float* __restrict__ partPos, float* __restrict__ partAll,
    int N, int tilesPerSlice)
{
    __shared__ _Float16 sbuf[4 * PLANE];          // 2 buffers x {hi,lo} planes

    const int tid  = threadIdx.x;
    const int lane = tid & 31;
    const int wv   = tid >> 5;
    const int hlf  = lane >> 4;
    const int ln   = lane & 15;

    const int rowBase = blockIdx.x * ROWS_PER_BLOCK + wv * TW;

    // A tile resident in VGPRs, already split; k order used here (B matches):
    // lane m=ln; 16 contiguous K at 16*hlf.
    v16h Ah[4], Al[4];
    {
        const _Float16* ah = Ehi + (size_t)(rowBase + ln) * DIM;
        const _Float16* al = Elo + (size_t)(rowBase + ln) * DIM;
#pragma unroll
        for (int c = 0; c < 4; ++c) {
            const int b1 = 32 * c + 16 * hlf;
            const int b2 = b1 + 8;
            v8h h0 = *(const v8h*)(ah + b1);
            v8h h1 = *(const v8h*)(ah + b2);
            Ah[c] = __builtin_shufflevector(h0, h1, 0,1,2,3,4,5,6,7,8,9,10,11,12,13,14,15);
            v8h l0 = *(const v8h*)(al + b1);
            v8h l1 = *(const v8h*)(al + b2);
            Al[c] = __builtin_shufflevector(l0, l1, 0,1,2,3,4,5,6,7,8,9,10,11,12,13,14,15);
        }
    }

    int rowLab[8];
#pragma unroll
    for (int v = 0; v < 8; ++v) rowLab[v] = labels[rowBase + 8 * hlf + v];

    float allS[8], posS[8];
#pragma unroll
    for (int v = 0; v < 8; ++v) { allS[v] = 0.f; posS[v] = 0.f; }

    const unsigned int ldsBase = (unsigned int)(uintptr_t)(&sbuf[0]);
    const int t0 = blockIdx.y * tilesPerSlice;
    const int t1 = t0 + tilesPerSlice;

    auto issue = [&](int t) {
        const unsigned int off = (unsigned int)((t & 1) * 2 * PLANE * 2);   // bytes
        tdm_load_tile(Ehi + (size_t)(t * TW) * DIM, ldsBase + off);
        tdm_load_tile(Elo + (size_t)(t * TW) * DIM, ldsBase + off + PLANE * 2);
    };

    if (wv == 0) issue(t0);

    for (int t = t0; t < t1; ++t) {
        if (wv == 0) __builtin_amdgcn_s_wait_tensorcnt(0);
        __syncthreads();                           // tile t staged; tile t-1 reads done
        if (wv == 0 && t + 1 < t1) issue(t + 1);   // prefetch into other buffer

        const int bufOff = (t & 1) * 2 * PLANE;    // halves
        const int colBase = t * TW;
        const int lbCol = labels[colBase + ln];

        v8f acc = {};
#pragma unroll
        for (int c = 0; c < 4; ++c) {
            // B k order used here (matches A): lane n=ln; 16 contiguous K at 16*hlf.
            const int kb = 32 * c + 16 * hlf;
            v16h bh = *(const v16h*)(&sbuf[bufOff + ln * LDS_STRIDE + kb]);
            v16h bl = *(const v16h*)(&sbuf[bufOff + PLANE + ln * LDS_STRIDE + kb]);
            acc = __builtin_amdgcn_wmma_f32_16x16x32_f16(
                false, Al[c], false, bh, (short)0, acc, false, false);
            acc = __builtin_amdgcn_wmma_f32_16x16x32_f16(
                false, Ah[c], false, bl, (short)0, acc, false, false);
            acc = __builtin_amdgcn_wmma_f32_16x16x32_f16(
                false, Ah[c], false, bh, (short)0, acc, false, false);
        }

        const int j = colBase + ln;
#pragma unroll
        for (int v = 0; v < 8; ++v) {
            const int i = rowBase + 8 * hlf + v;
            const float e  = __builtin_amdgcn_exp2f(acc[v] * SCALE_EXP2);
            const float em = (i != j) ? e : 0.f;                  // v_cndmask
            allS[v] += em;
            posS[v] += (rowLab[v] == lbCol) ? em : 0.f;           // v_cndmask
        }
    }

#pragma unroll
    for (int v = 0; v < 8; ++v) {
#pragma unroll
        for (int m = 8; m >= 1; m >>= 1) {
            allS[v] += __shfl_xor(allS[v], m, 16);
            posS[v] += __shfl_xor(posS[v], m, 16);
        }
    }
    if (ln == 0) {
#pragma unroll
        for (int v = 0; v < 8; ++v) {
            const int i = rowBase + 8 * hlf + v;
            partPos[(size_t)blockIdx.y * N + i] = posS[v];
            partAll[(size_t)blockIdx.y * N + i] = allS[v];
        }
    }
}

// ----------------------------------- fallback main (in-kernel split staging)
__global__ __launch_bounds__(256) void contrastive_fb(
    const float* __restrict__ emb, const int* __restrict__ labels,
    float* __restrict__ partPos, float* __restrict__ partAll,
    int N, int tilesPerSlice)
{
    __shared__ _Float16 lhi[PLANE];
    __shared__ _Float16 llo[PLANE];

    const int tid  = threadIdx.x;
    const int lane = tid & 31;
    const int wv   = tid >> 5;
    const int hlf  = lane >> 4;
    const int ln   = lane & 15;

    const int rowBase = blockIdx.x * ROWS_PER_BLOCK + wv * TW;

    v16h Ah[4], Al[4];
    {
        const float* arow = emb + (size_t)(rowBase + ln) * DIM;
#pragma unroll
        for (int c = 0; c < 4; ++c) {
            const int b1 = 32 * c + 16 * hlf;
            const int b2 = b1 + 8;
            float4 fa = ((const float4*)(arow + b1))[0];
            float4 fb = ((const float4*)(arow + b1))[1];
            float4 fc = ((const float4*)(arow + b2))[0];
            float4 fd = ((const float4*)(arow + b2))[1];
            float xs[16] = {fa.x, fa.y, fa.z, fa.w, fb.x, fb.y, fb.z, fb.w,
                            fc.x, fc.y, fc.z, fc.w, fd.x, fd.y, fd.z, fd.w};
#pragma unroll
            for (int q = 0; q < 16; ++q) {
                _Float16 h = (_Float16)xs[q];
                Ah[c][q] = h;
                Al[c][q] = (_Float16)(xs[q] - (float)h);
            }
        }
    }

    int rowLab[8];
#pragma unroll
    for (int v = 0; v < 8; ++v) rowLab[v] = labels[rowBase + 8 * hlf + v];

    float allS[8], posS[8];
#pragma unroll
    for (int v = 0; v < 8; ++v) { allS[v] = 0.f; posS[v] = 0.f; }

    const int t0 = blockIdx.y * tilesPerSlice;
    for (int t = t0; t < t0 + tilesPerSlice; ++t) {
        const int colBase = t * TW;

        __syncthreads();
        {
            const int r  = tid >> 4;
            const int kb = (tid & 15) << 3;
            const float* src = emb + (size_t)(colBase + r) * DIM + kb;
            float4 f0 = ((const float4*)src)[0];
            float4 f1 = ((const float4*)src)[1];
            float xs[8] = {f0.x, f0.y, f0.z, f0.w, f1.x, f1.y, f1.z, f1.w};
            v8h hv, lv;
#pragma unroll
            for (int q = 0; q < 8; ++q) {
                _Float16 h = (_Float16)xs[q];
                hv[q] = h;
                lv[q] = (_Float16)(xs[q] - (float)h);
            }
            *(v8h*)(&lhi[r * LDS_STRIDE + kb]) = hv;
            *(v8h*)(&llo[r * LDS_STRIDE + kb]) = lv;
        }
        __syncthreads();

        const int lbCol = labels[colBase + ln];
        v8f acc = {};
#pragma unroll
        for (int c = 0; c < 4; ++c) {
            const int kb = 32 * c + 16 * hlf;
            v16h bh = *(const v16h*)(&lhi[ln * LDS_STRIDE + kb]);
            v16h bl = *(const v16h*)(&llo[ln * LDS_STRIDE + kb]);
            acc = __builtin_amdgcn_wmma_f32_16x16x32_f16(
                false, Al[c], false, bh, (short)0, acc, false, false);
            acc = __builtin_amdgcn_wmma_f32_16x16x32_f16(
                false, Ah[c], false, bl, (short)0, acc, false, false);
            acc = __builtin_amdgcn_wmma_f32_16x16x32_f16(
                false, Ah[c], false, bh, (short)0, acc, false, false);
        }

        const int j = colBase + ln;
#pragma unroll
        for (int v = 0; v < 8; ++v) {
            const int i = rowBase + 8 * hlf + v;
            const float e  = __builtin_amdgcn_exp2f(acc[v] * SCALE_EXP2);
            const float em = (i != j) ? e : 0.f;
            allS[v] += em;
            posS[v] += (rowLab[v] == lbCol) ? em : 0.f;
        }
    }

#pragma unroll
    for (int v = 0; v < 8; ++v) {
#pragma unroll
        for (int m = 8; m >= 1; m >>= 1) {
            allS[v] += __shfl_xor(allS[v], m, 16);
            posS[v] += __shfl_xor(posS[v], m, 16);
        }
    }
    if (ln == 0) {
#pragma unroll
        for (int v = 0; v < 8; ++v) {
            const int i = rowBase + 8 * hlf + v;
            partPos[(size_t)blockIdx.y * N + i] = posS[v];
            partAll[(size_t)blockIdx.y * N + i] = allS[v];
        }
    }
}

// ------------------------------------------------------------------ finalize
__global__ __launch_bounds__(256) void contrastive_final(
    const float* __restrict__ partPos, const float* __restrict__ partAll,
    float* __restrict__ out, int N, int S)
{
    __shared__ float sL[256];
    __shared__ float sC[256];
    float ls = 0.f, cnt = 0.f;
    for (int i = threadIdx.x; i < N; i += 256) {
        float p = 0.f, a = 0.f;
        for (int s = 0; s < S; ++s) {
            p += partPos[(size_t)s * N + i];
            a += partAll[(size_t)s * N + i];
        }
        if (p > 0.f) {
            ls += __logf(a) - __logf(p);
            cnt += 1.f;
        }
    }
    sL[threadIdx.x] = ls;
    sC[threadIdx.x] = cnt;
    __syncthreads();
    for (int o = 128; o > 0; o >>= 1) {
        if (threadIdx.x < o) {
            sL[threadIdx.x] += sL[threadIdx.x + o];
            sC[threadIdx.x] += sC[threadIdx.x + o];
        }
        __syncthreads();
    }
    if (threadIdx.x == 0) out[0] = (sC[0] > 0.f) ? (sL[0] / sC[0]) : 0.f;
}

// -------------------------------------------------------------------- launch
extern "C" void kernel_launch(void* const* d_in, const int* in_sizes, int n_in,
                              void* d_out, int out_size, void* d_ws, size_t ws_size,
                              hipStream_t stream) {
    const float* emb    = (const float*)d_in[0];
    const int*   labels = (const int*)d_in[1];
    const int N  = in_sizes[1];               // 16384
    const int NT = N / TW;                    // 1024 column tiles

    const size_t splitBytes = 2ull * (size_t)N * DIM * sizeof(_Float16);  // hi+lo planes

    if (ws_size >= splitBytes + 2ull * (size_t)N * sizeof(float)) {
        // Fast path: pre-split planes + TDM staging.
        int S = 8;
        while (S > 1 && splitBytes + 2ull * (size_t)S * N * sizeof(float) > ws_size) S >>= 1;
        const int tilesPerSlice = NT / S;

        _Float16* Ehi = (_Float16*)d_ws;
        _Float16* Elo = Ehi + (size_t)N * DIM;
        float* partPos = (float*)((char*)d_ws + splitBytes);
        float* partAll = partPos + (size_t)S * N;

        const int nVec8 = N * DIM / 8;
        prep_split<<<(nVec8 + 255) / 256, 256, 0, stream>>>(emb, Ehi, Elo, nVec8);
        dim3 grid(N / ROWS_PER_BLOCK, S);
        contrastive_tdm<<<grid, 256, 0, stream>>>(Ehi, Elo, labels,
                                                  partPos, partAll, N, tilesPerSlice);
        contrastive_final<<<1, 256, 0, stream>>>(partPos, partAll, (float*)d_out, N, S);
    } else {
        // Fallback: in-kernel split staging.
        int S = 8;
        while (S > 1 && 2ull * (size_t)S * N * sizeof(float) > ws_size) S >>= 1;
        const int tilesPerSlice = NT / S;

        float* partPos = (float*)d_ws;
        float* partAll = partPos + (size_t)S * N;

        dim3 grid(N / ROWS_PER_BLOCK, S);
        contrastive_fb<<<grid, 256, 0, stream>>>(emb, labels,
                                                 partPos, partAll, N, tilesPerSlice);
        contrastive_final<<<1, 256, 0, stream>>>(partPos, partAll, (float*)d_out, N, S);
    }
}
